// DetModel_82832739270839
// MI455X (gfx1250) — matrix operand order repaired and checked by reading, not compiled, not run on hardware
//
#include <hip/hip_runtime.h>
#include <hip/hip_bf16.h>
#include <math.h>

typedef __attribute__((ext_vector_type(16))) _Float16 v16h;
typedef __attribute__((ext_vector_type(8)))  _Float16 h8;
typedef __attribute__((ext_vector_type(4)))  _Float16 h4;
typedef __attribute__((ext_vector_type(8)))  float    v8f;

union AF { v16h v; _Float16 h[16]; };
union A8 { h8   v; _Float16 h[8];  };
union H4 { h4   v; _Float16 h[4];  };
union CF { v8f  v; float    f[8];  };

// ---------------------------------------------------------------------------
// Weight pack: f32 HWIO [T*Cin, Cout] -> f16 WMMA B-fragment layout
// dst[((((cg*T + t)*KB + kb)*32 + lane)*16 + e)] = w[(t*Cin + kb*32 + g*8 + (e<8 ? e : e+8))*Cout + cg*16 + n]
// ---------------------------------------------------------------------------
__global__ void pack_w(const float* __restrict__ w, _Float16* __restrict__ dst,
                       int T, int Cin, int Cout) {
    size_t i = (size_t)blockIdx.x * 256 + threadIdx.x;
    size_t total = (size_t)T * Cin * Cout;
    if (i >= total) return;
    int e    = (int)(i & 15);
    int lane = (int)((i >> 4) & 31);
    size_t r = i >> 9;
    int KB = Cin >> 5;
    int kb = (int)(r % KB); r /= KB;
    int t  = (int)(r % T);
    int cg = (int)(r / T);
    int g = lane >> 4, nn = lane & 15;
    int cin = kb * 32 + g * 8 + (e < 8 ? e : e + 8);
    dst[i] = (_Float16)w[((size_t)t * Cin + cin) * Cout + cg * 16 + nn];
}

// reg_w [T*Cin,12] + wt_w [T*Cin,3] packed together into Cout=16 (ch15 = 0)
__global__ void pack_w2(const float* __restrict__ wr, const float* __restrict__ ww,
                        _Float16* __restrict__ dst, int T, int Cin) {
    size_t i = (size_t)blockIdx.x * 256 + threadIdx.x;
    size_t total = (size_t)T * Cin * 16;
    if (i >= total) return;
    int e    = (int)(i & 15);
    int lane = (int)((i >> 4) & 31);
    size_t r = i >> 9;
    int KB = Cin >> 5;
    int kb = (int)(r % KB); r /= KB;
    int t  = (int)(r % T);
    int g = lane >> 4, nn = lane & 15;       // cg == 0
    int cin = kb * 32 + g * 8 + (e < 8 ? e : e + 8);
    size_t row = (size_t)t * Cin + cin;
    float v = 0.f;
    if (nn < 12)      v = wr[row * 12 + nn];
    else if (nn < 15) v = ww[row * 3 + (nn - 12)];
    dst[i] = (_Float16)v;
}

__global__ void make_bias16(const float* __restrict__ rb, const float* __restrict__ wb,
                            float* __restrict__ out16) {
    int t = threadIdx.x;
    if (t < 12)      out16[t] = rb[t];
    else if (t < 15) out16[t] = wb[t - 12];
    else if (t < 16) out16[t] = 0.f;
}

// ---------------------------------------------------------------------------
// Generic 3x3/1x1 SAME conv (NHWC) as implicit GEMM with WMMA f16.
// One wave = 16 pixels (consecutive x in a row, y from blockIdx.y, image n from
// blockIdx.z) x (CGW * 16) output channels. A-fragment is built once per K-step
// and fed to CGW WMMAs. TIN/TOUT select f32 or f16 activation storage; f16
// storage is bit-identical for WMMA consumers (A is rounded to f16 anyway).
// Requires W/16 to be a power of two (true for all uses: W in {32, 256}).
// Also used as dense GEMM via H=1, KH=KW=1 (W = rows, Cin = K).
// ---------------------------------------------------------------------------
template<int CGW, typename TIN, typename TOUT>
__global__ __launch_bounds__(256) void conv_wmma(
    const TIN* __restrict__ in, const _Float16* __restrict__ wf,
    const float* __restrict__ bias, TOUT* __restrict__ out,
    int H, int W, int Cin, int Cout, int KH, int KW, int relu)
{
    const int nTX = W >> 4, nCG = Cout >> 4, KB = Cin >> 5, T = KH * KW;
    const int nChunk = (nCG + CGW - 1) / CGW;
    const int widRow = blockIdx.x * 8 + (threadIdx.x >> 5);
    if (widRow >= nTX * nChunk) return;             // wave-uniform exit

    const int txShift = 31 - __clz(nTX);            // nTX is a power of two
    const int tx    = widRow & (nTX - 1);
    const int chunk = widRow >> txShift;
    const int y = blockIdx.y, n = blockIdx.z;
    const int lane = threadIdx.x & 31;
    const int g = lane >> 4, mrow = lane & 15;
    const int x0 = tx << 4;
    const int ph = KH >> 1, pw = KW >> 1;
    const int cg0 = chunk * CGW;
    const int cgn = (nCG - cg0) < CGW ? (nCG - cg0) : CGW;

    v8f acc[CGW];
    #pragma unroll
    for (int ci = 0; ci < CGW; ++ci)
        acc[ci] = (v8f){0.f,0.f,0.f,0.f,0.f,0.f,0.f,0.f};

    const size_t cgStride = ((size_t)T * KB) << 9;  // halves per channel-group
    const _Float16* wfl = wf + (size_t)cg0 * cgStride + ((size_t)lane << 4);

    for (int ky = 0; ky < KH; ++ky) {
        const int yin = y + ky - ph;
        const bool vy = (yin >= 0) & (yin < H);
        for (int kx = 0; kx < KW; ++kx) {
            const int t = ky * KW + kx;
            const int xin = x0 + mrow + kx - pw;
            const bool vx = vy & (xin >= 0) & (xin < W);
            const TIN* abase = in + (((size_t)n * H + (vy ? yin : 0)) * W + (vx ? xin : 0)) * Cin;
            const size_t tkBase = ((size_t)t * KB) << 9;
            for (int kb = 0; kb < KB; ++kb) {
                __builtin_prefetch(wfl + tkBase + ((size_t)(kb + 1) << 9), 0, 0);
                AF a;
                if (vx) {
                    const TIN* ap = abase + (kb << 5) + (g << 3);
                    if constexpr (sizeof(TIN) == 4) {
                        float4 q0 = *(const float4*)(ap);
                        float4 q1 = *(const float4*)(ap + 4);
                        float4 q2 = *(const float4*)(ap + 16);
                        float4 q3 = *(const float4*)(ap + 20);
                        a.h[0]=(_Float16)q0.x;  a.h[1]=(_Float16)q0.y;  a.h[2]=(_Float16)q0.z;  a.h[3]=(_Float16)q0.w;
                        a.h[4]=(_Float16)q1.x;  a.h[5]=(_Float16)q1.y;  a.h[6]=(_Float16)q1.z;  a.h[7]=(_Float16)q1.w;
                        a.h[8]=(_Float16)q2.x;  a.h[9]=(_Float16)q2.y;  a.h[10]=(_Float16)q2.z; a.h[11]=(_Float16)q2.w;
                        a.h[12]=(_Float16)q3.x; a.h[13]=(_Float16)q3.y; a.h[14]=(_Float16)q3.z; a.h[15]=(_Float16)q3.w;
                    } else {
                        A8 q0, q1;
                        q0.v = *(const h8*)(ap);
                        q1.v = *(const h8*)(ap + 16);
                        #pragma unroll
                        for (int e = 0; e < 8; ++e) { a.h[e] = q0.h[e]; a.h[8 + e] = q1.h[e]; }
                    }
                } else {
                    #pragma unroll
                    for (int e = 0; e < 16; ++e) a.h[e] = (_Float16)0.f;
                }
                const size_t tk = tkBase + ((size_t)kb << 9);
                #pragma unroll
                for (int ci = 0; ci < CGW; ++ci) {
                    if (ci < cgn) {                 // wave-uniform predicate
                        v16h bf = *(const v16h*)(wfl + (size_t)ci * cgStride + tk);
                        acc[ci] = __builtin_amdgcn_wmma_f32_16x16x32_f16(
                                      false, a.v, false, bf, (short)0, acc[ci], false, false);
                    }
                }
            }
        }
    }

    TOUT* orow = out + (((size_t)n * H + y) * W + x0) * Cout + mrow;
    #pragma unroll
    for (int ci = 0; ci < CGW; ++ci) {
        if (ci < cgn) {
            const int ch = ((cg0 + ci) << 4);
            const float bv = bias[ch + mrow];       // lane's D column = lane&15
            CF cf; cf.v = acc[ci];
            TOUT* obase = orow + ch;
            #pragma unroll
            for (int rr = 0; rr < 8; ++rr) {
                float v = cf.f[rr] + bv;
                if (relu) v = fmaxf(v, 0.f);
                obase[(size_t)(rr + (g << 3)) * Cout] = (TOUT)v;   // D row M = rr + 8*g
            }
        }
    }
}

// ---------------------------------------------------------------------------
// Classification head: mean(feat5) -> fc1 relu -> fc2 relu -> fc3 -> softmax/argmax
// one block per batch, 256 threads
// ---------------------------------------------------------------------------
__global__ void cls_head(const float* __restrict__ feat5,
                         const float* __restrict__ w1, const float* __restrict__ b1,
                         const float* __restrict__ w2, const float* __restrict__ b2,
                         const float* __restrict__ w3, const float* __restrict__ b3,
                         float* __restrict__ pred_out, int* __restrict__ cls_out) {
    __shared__ float g[256], h1[256], h2[256], lg[3];
    const int b = blockIdx.x, t = threadIdx.x;
    const float* f = feat5 + (size_t)b * 256 * 256;
    float s = 0.f;
    for (int p = 0; p < 256; ++p) s += f[(size_t)p * 256 + t];
    g[t] = s * (1.f / 256.f);
    __syncthreads();
    float a = b1[t];
    for (int k = 0; k < 256; ++k) a += g[k] * w1[(size_t)k * 256 + t];
    h1[t] = fmaxf(a, 0.f);
    __syncthreads();
    a = b2[t];
    for (int k = 0; k < 256; ++k) a += h1[k] * w2[(size_t)k * 256 + t];
    h2[t] = fmaxf(a, 0.f);
    __syncthreads();
    if (t < 3) {
        float z = b3[t];
        for (int k = 0; k < 256; ++k) z += h2[k] * w3[(size_t)k * 3 + t];
        lg[t] = z;
    }
    __syncthreads();
    if (t == 0) {
        float mx = fmaxf(lg[0], fmaxf(lg[1], lg[2]));
        float e0 = expf(lg[0] - mx), e1 = expf(lg[1] - mx), e2 = expf(lg[2] - mx);
        float inv = 1.f / (e0 + e1 + e2);
        pred_out[b * 3 + 0] = e0 * inv;
        pred_out[b * 3 + 1] = e1 * inv;
        pred_out[b * 3 + 2] = e2 * inv;
        int c = 0; float best = lg[0];
        if (lg[1] > best) { best = lg[1]; c = 1; }
        if (lg[2] > best) { best = lg[2]; c = 2; }
        cls_out[b] = c;
    }
}

// per-class gather of reg/wt channels. i over B*H*W (H*W = 65536)
__global__ void gather_regwt(const float* __restrict__ regwt, const int* __restrict__ cls,
                             float* __restrict__ off, float* __restrict__ sz,
                             float* __restrict__ wt) {
    int i = blockIdx.x * 256 + threadIdx.x;
    int c = cls[i >> 16];
    const float* p = regwt + (size_t)i * 16;
    off[(size_t)i * 2 + 0] = p[4 * c + 0];
    off[(size_t)i * 2 + 1] = p[4 * c + 1];
    sz [(size_t)i * 2 + 0] = p[4 * c + 2];
    sz [(size_t)i * 2 + 1] = p[4 * c + 3];
    wt[i] = p[12 + c];
}

// bilinear crop: feat0 [4,512,512,64], boxes [4,64,4] -> crops [256,32,32,64] f16
// one 4-channel group per thread; idx = ((n*32+i)*32+j)*16 + cq
__global__ void crop_kernel(const float* __restrict__ feat0, const float* __restrict__ boxes,
                            _Float16* __restrict__ crops) {
    int idx = blockIdx.x * 256 + threadIdx.x;
    int cq = idx & 15, j = (idx >> 4) & 31, i = (idx >> 9) & 31, n = idx >> 14;
    int b = n >> 6;
    const float* bx = boxes + (size_t)n * 4;
    float y1 = bx[0], x1 = bx[1], y2 = bx[2], x2 = bx[3];
    float ys = (y1 + (float)i * (1.f / 31.f) * (y2 - y1)) * 511.f;
    float xs = (x1 + (float)j * (1.f / 31.f) * (x2 - x1)) * 511.f;
    int y0 = (int)floorf(ys); y0 = y0 < 0 ? 0 : (y0 > 510 ? 510 : y0);
    int x0 = (int)floorf(xs); x0 = x0 < 0 ? 0 : (x0 > 510 ? 510 : x0);
    float wy = ys - (float)y0, wx = xs - (float)x0;
    const float* fb = feat0 + (size_t)b * 512 * 512 * 64;
    size_t p00 = ((size_t)y0 * 512 + x0) * 64 + cq * 4;
    float4 v00 = *(const float4*)(fb + p00);
    float4 v01 = *(const float4*)(fb + p00 + 64);
    float4 v10 = *(const float4*)(fb + p00 + 512 * 64);
    float4 v11 = *(const float4*)(fb + p00 + 512 * 64 + 64);
    H4 o;
    o.h[0] = (_Float16)((1.f - wy) * ((1.f - wx) * v00.x + wx * v01.x) + wy * ((1.f - wx) * v10.x + wx * v11.x));
    o.h[1] = (_Float16)((1.f - wy) * ((1.f - wx) * v00.y + wx * v01.y) + wy * ((1.f - wx) * v10.y + wx * v11.y));
    o.h[2] = (_Float16)((1.f - wy) * ((1.f - wx) * v00.z + wx * v01.z) + wy * ((1.f - wx) * v10.z + wx * v11.z));
    o.h[3] = (_Float16)((1.f - wy) * ((1.f - wx) * v00.w + wx * v01.w) + wy * ((1.f - wx) * v10.w + wx * v11.w));
    *(h4*)(crops + (size_t)idx * 4) = o.v;
}

// seg conv 96->1 (selected class only), i over N*32*32
__global__ void seg_head(const float* __restrict__ s, const float* __restrict__ so_w,
                         const float* __restrict__ so_b, const int* __restrict__ cls,
                         float* __restrict__ out) {
    int i = blockIdx.x * 256 + threadIdx.x;
    int n = i >> 10, y = (i >> 5) & 31, x = i & 31;
    int c = cls[n >> 6];
    float acc = so_b[c];
    for (int ky = 0; ky < 3; ++ky) {
        int yy = y + ky - 1; if (yy < 0 || yy >= 32) continue;
        for (int kx = 0; kx < 3; ++kx) {
            int xx = x + kx - 1; if (xx < 0 || xx >= 32) continue;
            const float* sp = s + (((size_t)n * 32 + yy) * 32 + xx) * 96;
            const float* wp = so_w + (size_t)((ky * 3 + kx) * 96) * 3 + c;
            for (int ci = 0; ci < 96; ++ci) acc += sp[ci] * wp[(size_t)ci * 3];
        }
    }
    out[i] = acc;
}

// 4x4 avg pool: s [256,32,32,96] f32 -> pooled [256,6144] f16; i over 256*8*8*96
__global__ void pool_kernel(const float* __restrict__ s, _Float16* __restrict__ pooled) {
    int i = blockIdx.x * 256 + threadIdx.x;
    int c = i % 96; int t = i / 96;
    int wx = t & 7; t >>= 3;
    int hy = t & 7; int n = t >> 3;
    float a = 0.f;
    for (int dy = 0; dy < 4; ++dy)
        for (int dx = 0; dx < 4; ++dx)
            a += s[(((size_t)n * 32 + hy * 4 + dy) * 32 + wx * 4 + dx) * 96 + c];
    pooled[i] = (_Float16)(a * (1.f / 16.f));
}

// final dense 256->3 with class select; 256 threads, one per ROI
__global__ void score3(const float* __restrict__ h2, const float* __restrict__ w3,
                       const float* __restrict__ b3, const int* __restrict__ cls,
                       float* __restrict__ out) {
    int n = threadIdx.x;
    int c = cls[n >> 6];
    float a = b3[c];
    const float* hp = h2 + (size_t)n * 256;
    for (int k = 0; k < 256; ++k) a += hp[k] * w3[(size_t)k * 3 + c];
    out[n] = a;
}

// ---------------------------------------------------------------------------
extern "C" void kernel_launch(void* const* d_in, const int* in_sizes, int n_in,
                              void* d_out, int out_size, void* d_ws, size_t ws_size,
                              hipStream_t stream) {
    const float* feat0 = (const float*)d_in[0];
    const float* feat1 = (const float*)d_in[1];
    const float* feat5 = (const float*)d_in[2];
    const float* boxes = (const float*)d_in[3];
    const float* cb1_w = (const float*)d_in[4];  const float* cb1_b = (const float*)d_in[5];
    const float* cb2_w = (const float*)d_in[6];  const float* cb2_b = (const float*)d_in[7];
    const float* reg_w = (const float*)d_in[8];
    const float* reg_b = (const float*)d_in[9];
    const float* wt_w  = (const float*)d_in[10]; const float* wt_b  = (const float*)d_in[11];
    const float* fc1_w = (const float*)d_in[12]; const float* fc1_b = (const float*)d_in[13];
    const float* fc2_w = (const float*)d_in[14]; const float* fc2_b = (const float*)d_in[15];
    const float* fc3_w = (const float*)d_in[16]; const float* fc3_b = (const float*)d_in[17];
    const float* sc1_w = (const float*)d_in[18]; const float* sc1_b = (const float*)d_in[19];
    const float* sc2_w = (const float*)d_in[20]; const float* sc2_b = (const float*)d_in[21];
    const float* so_w  = (const float*)d_in[22]; const float* so_b  = (const float*)d_in[23];
    const float* sd1_w = (const float*)d_in[24]; const float* sd1_b = (const float*)d_in[25];
    const float* sd2_w = (const float*)d_in[26]; const float* sd2_b = (const float*)d_in[27];
    const float* sd3_w = (const float*)d_in[28]; const float* sd3_b = (const float*)d_in[29];

    // ---- output sections (f32, concatenated in return order) ----
    float* out_f = (float*)d_out;
    float* o_off   = out_f;                 // [4,256,256,2]
    float* o_sz    = out_f + 524288;        // [4,256,256,2]
    float* o_wt    = out_f + 1048576;       // [4,256,256,1]
    float* o_pred  = out_f + 1310720;       // [4,3]
    float* o_seg   = out_f + 1310732;       // [256,32,32,1]
    float* o_score = out_f + 1572876;       // [256,1]

    // ---- workspace layout ----
    char* ws = (char*)d_ws;
    size_t off = 0;
    auto alloc = [&](size_t bytes) -> char* {
        char* p = ws + off;
        off = (off + bytes + 255) & ~(size_t)255;
        return p;
    };
    _Float16* x1h    = (_Float16*)alloc((size_t)4 * 256 * 256 * 256 * 2);
    _Float16* x2h    = (_Float16*)alloc((size_t)4 * 256 * 256 * 256 * 2);
    float*    regwt  = (float*)   alloc((size_t)4 * 256 * 256 * 16 * 4);
    _Float16* crops  = (_Float16*)alloc((size_t)256 * 32 * 32 * 64 * 2);
    _Float16* s1h    = (_Float16*)alloc((size_t)256 * 32 * 32 * 96 * 2);
    float*    s2     = (float*)   alloc((size_t)256 * 32 * 32 * 96 * 4);
    _Float16* pooled = (_Float16*)alloc((size_t)256 * 6144 * 2);
    _Float16* h1h    = (_Float16*)alloc((size_t)256 * 256 * 2);
    float*    h2     = (float*)   alloc((size_t)256 * 256 * 4);
    int*      cls    = (int*)     alloc(4 * sizeof(int));
    float*    bias16 = (float*)   alloc(16 * sizeof(float));
    _Float16* wf_cb1 = (_Float16*)alloc((size_t)9 * 128 * 256 * 2);
    _Float16* wf_cb2 = (_Float16*)alloc((size_t)9 * 256 * 256 * 2);
    _Float16* wf_rw  = (_Float16*)alloc((size_t)9 * 256 * 16 * 2);
    _Float16* wf_sc1 = (_Float16*)alloc((size_t)9 * 64 * 96 * 2);
    _Float16* wf_sc2 = (_Float16*)alloc((size_t)9 * 96 * 96 * 2);
    _Float16* wf_sd1 = (_Float16*)alloc((size_t)6144 * 256 * 2);
    _Float16* wf_sd2 = (_Float16*)alloc((size_t)256 * 256 * 2);
    (void)ws_size; (void)in_sizes; (void)n_in; (void)out_size;

    auto cdiv = [](long a, long b) { return (a + b - 1) / b; };
    // blocks along x for a conv: ceil(nTX * nChunk / 8) with 8 waves per block
    auto xblk = [&](int W, int Cout, int CGW) {
        int nTX = W >> 4, nCG = Cout >> 4;
        int nChunk = (nCG + CGW - 1) / CGW;
        return (unsigned)cdiv((long)nTX * nChunk, 8);
    };

    // ---- 1) pack weights to f16 WMMA fragments ----
    pack_w<<<cdiv((long)9*128*256,256),256,0,stream>>>(cb1_w, wf_cb1, 9, 128, 256);
    pack_w<<<cdiv((long)9*256*256,256),256,0,stream>>>(cb2_w, wf_cb2, 9, 256, 256);
    pack_w<<<cdiv((long)9*64*96,  256),256,0,stream>>>(sc1_w, wf_sc1, 9, 64, 96);
    pack_w<<<cdiv((long)9*96*96,  256),256,0,stream>>>(sc2_w, wf_sc2, 9, 96, 96);
    pack_w<<<cdiv((long)6144*256, 256),256,0,stream>>>(sd1_w, wf_sd1, 1, 6144, 256);
    pack_w<<<cdiv((long)256*256,  256),256,0,stream>>>(sd2_w, wf_sd2, 1, 256, 256);
    pack_w2<<<cdiv((long)9*256*16,256),256,0,stream>>>(reg_w, wt_w, wf_rw, 9, 256);
    make_bias16<<<1, 16, 0, stream>>>(reg_b, wt_b, bias16);

    // ---- 2) classification head ----
    cls_head<<<4, 256, 0, stream>>>(feat5, fc1_w, fc1_b, fc2_w, fc2_b, fc3_w, fc3_b,
                                    o_pred, cls);

    // ---- 3) regression branch: cb1 -> cb2 -> fused reg/wt conv -> gather ----
    conv_wmma<4, float, _Float16><<<dim3(xblk(256,256,4),256,4),256,0,stream>>>(
        feat1, wf_cb1, cb1_b, x1h, 256,256,128,256, 3,3, 1);
    conv_wmma<4, _Float16, _Float16><<<dim3(xblk(256,256,4),256,4),256,0,stream>>>(
        x1h, wf_cb2, cb2_b, x2h, 256,256,256,256, 3,3, 1);
    conv_wmma<1, _Float16, float><<<dim3(xblk(256,16,1),256,4),256,0,stream>>>(
        x2h, wf_rw, bias16, regwt, 256,256,256,16, 3,3, 0);
    gather_regwt<<<(4*256*256)/256, 256, 0, stream>>>(regwt, cls, o_off, o_sz, o_wt);

    // ---- 4) ROI crops + segmentation convs ----
    crop_kernel<<<(256*32*32*16)/256, 256, 0, stream>>>(feat0, boxes, crops);
    conv_wmma<2, _Float16, _Float16><<<dim3(xblk(32,96,2),32,256),256,0,stream>>>(
        crops, wf_sc1, sc1_b, s1h, 32,32,64,96, 3,3, 1);
    conv_wmma<2, _Float16, float><<<dim3(xblk(32,96,2),32,256),256,0,stream>>>(
        s1h, wf_sc2, sc2_b, s2, 32,32,96,96, 3,3, 1);
    seg_head<<<(256*32*32)/256, 256, 0, stream>>>(s2, so_w, so_b, cls, o_seg);

    // ---- 5) score head: pool -> sd1 (WMMA GEMM) -> sd2 (WMMA GEMM) -> sd3+select ----
    pool_kernel<<<(256*8*8*96)/256, 256, 0, stream>>>(s2, pooled);
    conv_wmma<4, _Float16, _Float16><<<dim3(xblk(256,256,4),1,1),256,0,stream>>>(
        pooled, wf_sd1, sd1_b, h1h, 1,256,6144,256, 1,1, 1);
    conv_wmma<4, _Float16, float><<<dim3(xblk(256,256,4),1,1),256,0,stream>>>(
        h1h, wf_sd2, sd2_b, h2, 1,256,256,256, 1,1, 1);
    score3<<<1, 256, 0, stream>>>(h2, sd3_w, sd3_b, cls, o_score);
}
